// DifferentialCausalAttention_2680059593233
// MI455X (gfx1250) — matrix operand order repaired and checked by reading, not compiled, not run on hardware
//
#include <hip/hip_runtime.h>

// ---------------------------------------------------------------------------
// Types for CDNA5 WMMA (wave32) and TDM descriptors
// ---------------------------------------------------------------------------
typedef __attribute__((ext_vector_type(16))) __bf16 v16bf;
typedef __attribute__((ext_vector_type(8)))  float  v8f;
typedef unsigned int v4u __attribute__((ext_vector_type(4)));
typedef int          v4i __attribute__((ext_vector_type(4)));
typedef int          v8i __attribute__((ext_vector_type(8)));

union Frag {
    uint4 u[2];
    v16bf v;
};

__device__ __forceinline__ v8f wmma_bf16(const Frag& a, const Frag& b, v8f c) {
    return __builtin_amdgcn_wmma_f32_16x16x32_bf16(
        /*neg_a=*/false, a.v, /*neg_b=*/false, b.v,
        /*c_mod=*/(short)0, c, /*reuse_a=*/false, /*reuse_b=*/false);
}

// ---------------------------------------------------------------------------
// Tensor Data Mover: 2D tile load Global -> LDS.
// D# group0: count=1, lds_addr, 57b global_addr, type=2.
// D# group1: data_size=8B, pad_enable, pad_interval/pad_amount codes,
//            tensor_dim0/1, tile_dim0 (qwords/row), tile_dim1 (rows),
//            tensor_dim0_stride (qwords).
// Issued by one wave (TDM ignores EXEC); completion via TENSORcnt.
// ---------------------------------------------------------------------------
__device__ __forceinline__ void tdm_load_2d(const void* gaddr, unsigned lds_off,
                                            unsigned tile_qw, unsigned rows,
                                            unsigned stride_qw,
                                            unsigned pad_int_code,
                                            unsigned pad_amt_code) {
    unsigned long long ga = (unsigned long long)gaddr;
    v4u g0;
    g0.x = 1u;                                   // count=1 (valid descriptor)
    g0.y = lds_off;                              // LDS byte address
    g0.z = (unsigned)ga;                         // global_addr[31:0]
    g0.w = ((unsigned)(ga >> 32) & 0x01FFFFFFu)  // global_addr[56:32]
           | (2u << 30);                         // type=2 ("image")
    unsigned w0 = (3u << 16)                     // data_size = 8 bytes
                | (1u << 20)                     // pad_enable
                | (pad_int_code << 22)           // pad every 2^c qwords
                | (pad_amt_code << 25);          // pad (c+1) dwords
    unsigned td0 = stride_qw;                    // tensor_dim0 (no OOB needed)
    unsigned td1 = 0x00100000u;                  // tensor_dim1: large
    v8i g1;
    g1[0] = (int)w0;
    g1[1] = (int)((td0 & 0xFFFFu) << 16);                     // [63:48]=td0.lo
    g1[2] = (int)((td0 >> 16) | ((td1 & 0xFFFFu) << 16));     // td0.hi | td1.lo
    g1[3] = (int)((td1 >> 16) | (tile_qw << 16));             // td1.hi | tile_dim0
    g1[4] = (int)(rows & 0xFFFFu);                            // tile_dim1 (tile_dim2=0)
    g1[5] = (int)stride_qw;                                   // tensor_dim0_stride lo
    g1[6] = 0;
    g1[7] = 0;
    v4i z4 = {0, 0, 0, 0};
#if defined(__clang_major__) && (__clang_major__ >= 23)
    v8i z8 = {0, 0, 0, 0, 0, 0, 0, 0};
    __builtin_amdgcn_tensor_load_to_lds(g0, g1, z4, z4, z8, 0);
#else
    __builtin_amdgcn_tensor_load_to_lds(g0, g1, z4, z4, 0);
#endif
}

__device__ __forceinline__ unsigned lds_addr_of(const void* p) {
    return (unsigned)(unsigned long long)p;   // low 32 bits of flat = LDS offset
}

// ---------------------------------------------------------------------------
// Elementwise fp32 -> bf16 conversion
// ---------------------------------------------------------------------------
__global__ __launch_bounds__(256) void cvt_bf16_kernel(const float* __restrict__ src,
                                                       __bf16* __restrict__ dst,
                                                       size_t n) {
    size_t i = (size_t)blockIdx.x * blockDim.x + threadIdx.x;
    if (i < n) dst[i] = (__bf16)src[i];
}

// ---------------------------------------------------------------------------
// Fused convert + transpose: W[K][N] fp32 -> Wt[N][K] bf16 (32x32 LDS tiles)
// ---------------------------------------------------------------------------
__global__ __launch_bounds__(256) void cvtT_kernel(const float* __restrict__ src,
                                                   __bf16* __restrict__ dst,
                                                   int K, int N) {
    __shared__ float tile[32][33];
    int n0 = blockIdx.x * 32, k0 = blockIdx.y * 32;
    int tx = threadIdx.x & 31, ty = threadIdx.x >> 5;   // 32 x 8
#pragma unroll
    for (int i = ty; i < 32; i += 8)
        tile[i][tx] = src[(size_t)(k0 + i) * N + n0 + tx];
    __syncthreads();
#pragma unroll
    for (int i = ty; i < 32; i += 8)
        dst[(size_t)(n0 + i) * K + k0 + tx] = (__bf16)tile[tx][i];
}

// ---------------------------------------------------------------------------
// RoPE cos/sin tables, L x dh fp32
// ---------------------------------------------------------------------------
__global__ __launch_bounds__(256) void rope_table_kernel(float* __restrict__ cosT,
                                                         float* __restrict__ sinT,
                                                         int L, int dh) {
    int i = blockIdx.x * blockDim.x + threadIdx.x;
    if (i >= L * dh) return;
    int l = i / dh;
    int d = i % dh;
    int half = dh >> 1;
    int fi = d % half;
    float inv = powf(10000.0f, -2.0f * (float)fi / (float)dh);
    float fr = (float)l * inv;
    cosT[i] = cosf(fr);
    sinT[i] = sinf(fr);
}

// ---------------------------------------------------------------------------
// lam = sigmoid(x @ Wl + bl), stored [b][h][l] fp32.  One block per row of x.
// ---------------------------------------------------------------------------
__global__ __launch_bounds__(256) void lam_kernel(const float* __restrict__ x,
                                                  const float* __restrict__ Wl,
                                                  const float* __restrict__ bl,
                                                  float* __restrict__ lam,
                                                  int L, int D, int NH) {
    __shared__ float red[256];
    int row = blockIdx.x;             // b*L + l
    int t = threadIdx.x;
    int h = t & 15;                   // NH == 16
    int chunk = t >> 4;               // 16 chunks of 128
    const float* xr = x + (size_t)row * D + chunk * 128;
    const float* wr = Wl + (size_t)chunk * 128 * NH + h;
    float s = 0.f;
#pragma unroll 4
    for (int k = 0; k < 128; ++k) s += xr[k] * wr[k * NH];
    red[t] = s;
    __syncthreads();
    for (int st = 128; st >= 16; st >>= 1) {
        if (t < st) red[t] += red[t + st];
        __syncthreads();
    }
    if (t < NH) {
        float v = red[t] + bl[t];
        float sg = 1.0f / (1.0f + __expf(-v));
        int b = row / L, l = row % L;
        lam[((size_t)b * NH + t) * L + l] = sg;
    }
}

// ---------------------------------------------------------------------------
// Tiled WMMA GEMM:  C(MxN) = A(MxK bf16 row-major) * Wt^T  (Wt is N x K bf16).
// Block tile 128x128, k-step 32, TDM double-buffered LDS (wave 0 issues DMA,
// next slice prefetch overlaps the 8 WMMAs/wave of the current slice).
// Epilogues: PLAIN fp32 store | ROPE bf16 [b][head][l][d] | VT bf16 [b][head][d][l]
// ---------------------------------------------------------------------------
enum { EPI_PLAIN = 0, EPI_ROPE = 1, EPI_VT = 2 };

template <int EPI>
__global__ __launch_bounds__(256)
void gemm_bf16_wmma(const __bf16* __restrict__ A, const __bf16* __restrict__ Wt,
                    int M, int N, int K,
                    float* __restrict__ Cout, __bf16* __restrict__ Obf,
                    const float* __restrict__ cosT, const float* __restrict__ sinT,
                    int L) {
    constexpr int AS = 40;   // 64B row + 16B TDM pad = 80B stride
    __shared__ __align__(16) __bf16 As[2][128][AS];
    __shared__ __align__(16) __bf16 Bs[2][128][AS];

    const int t = threadIdx.x;
    const int wid = t >> 5, lane = t & 31;
    const int m = lane & 15, hi = lane >> 4;
    const int row0 = blockIdx.y * 128;
    const int n0 = blockIdx.x * 128;

    v8f acc[8];
#pragma unroll
    for (int j = 0; j < 8; ++j)
#pragma unroll
        for (int r = 0; r < 8; ++r) acc[j][r] = 0.f;

    const bool issuer = (__builtin_amdgcn_readfirstlane(wid) == 0);
    const __bf16* Abase = A + (size_t)row0 * K;
    const __bf16* Bbase = Wt + (size_t)n0 * K;
    const unsigned strideQW = (unsigned)(K >> 2);   // row stride in 8B units

    if (issuer) {
        tdm_load_2d(Abase, lds_addr_of(&As[0][0][0]), 8, 128, strideQW, 3, 3);
        tdm_load_2d(Bbase, lds_addr_of(&Bs[0][0][0]), 8, 128, strideQW, 3, 3);
    }
    int cur = 0;
    for (int k0 = 0; k0 < K; k0 += 32) {
        if (issuer) __builtin_amdgcn_s_wait_tensorcnt(0);
        __syncthreads();   // current buffers ready for all waves
        if (issuer && (k0 + 32) < K) {   // prefetch next slice into other buffer
            tdm_load_2d(Abase + k0 + 32, lds_addr_of(&As[cur ^ 1][0][0]), 8, 128,
                        strideQW, 3, 3);
            tdm_load_2d(Bbase + k0 + 32, lds_addr_of(&Bs[cur ^ 1][0][0]), 8, 128,
                        strideQW, 3, 3);
        }
        // A fragment: vec[e] = A[m][e + 8*hi] (e<8), A[m][e + 8 + 8*hi] (e>=8)
        Frag af;
        af.u[0] = *((const uint4*)&As[cur][16 * wid + m][8 * hi]);
        af.u[1] = *((const uint4*)&As[cur][16 * wid + m][16 + 8 * hi]);
#pragma unroll
        for (int j = 0; j < 8; ++j) {
            // B fragment: vec[e] = Wt[16*j+m][k0 + e + 8*hi] (e<8), Wt[16*j+m][k0 + e + 8 + 8*hi] (e>=8)
            Frag bf;
            bf.u[0] = *((const uint4*)&Bs[cur][16 * j + m][8 * hi]);
            bf.u[1] = *((const uint4*)&Bs[cur][16 * j + m][16 + 8 * hi]);
            acc[j] = wmma_bf16(af, bf, acc[j]);
        }
        cur ^= 1;
    }

    // ---------------- epilogues ----------------
    if (EPI == EPI_PLAIN) {
#pragma unroll
        for (int j = 0; j < 8; ++j)
#pragma unroll
            for (int r = 0; r < 8; ++r) {
                int grow = row0 + 16 * wid + r + 8 * hi;
                int gcol = n0 + 16 * j + m;
                Cout[(size_t)grow * N + gcol] = acc[j][r];
            }
    } else if (EPI == EPI_ROPE) {
        const int head = n0 >> 7;
        const int Htot = N >> 7;
#pragma unroll
        for (int j = 0; j < 4; ++j)
#pragma unroll
            for (int r = 0; r < 8; ++r) {
                int grow = row0 + 16 * wid + r + 8 * hi;
                int b = grow / L, l = grow % L;
                int dlo = 16 * j + m;
                float c0 = cosT[l * 128 + dlo];
                float s0 = sinT[l * 128 + dlo];
                float c1 = cosT[l * 128 + dlo + 64];
                float s1 = sinT[l * 128 + dlo + 64];
                float q0 = acc[j][r], q1 = acc[j + 4][r];
                float o0 = q0 * c0 - q1 * s0;   // d < 64 : rot = -t2
                float o1 = q1 * c1 + q0 * s1;   // d >= 64: rot = +t1
                size_t base = (((size_t)b * Htot + head) * (size_t)L + l) * 128;
                Obf[base + dlo] = (__bf16)o0;
                Obf[base + dlo + 64] = (__bf16)o1;
            }
    } else {  // EPI_VT
        const int head = n0 >> 7;
        const int Htot = N >> 7;
#pragma unroll
        for (int j = 0; j < 8; ++j) {
            int d = 16 * j + m;
            union { uint4 u; __bf16 h[8]; } pk;
#pragma unroll
            for (int r = 0; r < 8; ++r) pk.h[r] = (__bf16)acc[j][r];
            int grow0 = row0 + 16 * wid + 8 * hi;   // 8 consecutive rows
            int b = grow0 / L, l = grow0 % L;
            size_t base = (((size_t)b * Htot + head) * 128 + d) * (size_t)L + l;
            *((uint4*)&Obf[base]) = pk.u;
        }
    }
}

// ---------------------------------------------------------------------------
// Differential causal flash-attention.
// grid.x = L/64 query tiles, grid.y = B*NH.  256 threads = 8 waves:
// waves 0..3 -> p=0 rows [qt*64 .. +64), waves 4..7 -> p=1 same rows.
// K/V tiles staged by TDM (wave 0), scores & ctx via bf16 WMMA.
// ---------------------------------------------------------------------------
__global__ __launch_bounds__(256)
void attn_kernel(const __bf16* __restrict__ Qr,   // [b][2h+p][l][128]
                 const __bf16* __restrict__ Kr,   // [b][h][l][128]
                 const __bf16* __restrict__ Vt,   // [b][h][d][l]
                 const float* __restrict__ lam,   // [b][h][l]
                 __bf16* __restrict__ Aout,       // [b][l][h*128 + d]
                 int L) {
    constexpr int DH = 128, NH = 16;
    constexpr int KSTR = 136;   // 256B row + 16B TDM pad
    constexpr int VSTR = 40;    // 64B row + 16B TDM pad
    constexpr int PSTR = 40;
    __shared__ __align__(16) char smem[32768];
    __bf16* Ksm = (__bf16*)smem;                                    // [32][136]
    __bf16* Vsm = (__bf16*)(smem + 32 * KSTR * 2);                  // [128][40]
    __bf16* Psm = (__bf16*)(smem + 32 * KSTR * 2 + 128 * VSTR * 2); // 8x[16][40]
    float* Csm = (float*)smem;                                      // [64][128]

    const int t = threadIdx.x;
    const int wid = t >> 5, lane = t & 31;
    const int m = lane & 15, hi = lane >> 4;
    const int p = wid >> 2, wsub = wid & 3;
    const int bh = blockIdx.y;
    const int b = bh >> 4, h = bh & 15;
    const int qt = blockIdx.x;
    const int qb = qt * 64 + wsub * 16;        // wave's first query row
    const float scale = 0.08838834764831845f;  // 1/sqrt(128)
    const bool issuer = (__builtin_amdgcn_readfirstlane(wid) == 0);

    // Q fragments (4 chunks of K=32 over dh) straight from global
    const __bf16* qg = Qr + (((size_t)b * (2 * NH) + 2 * h + p) * (size_t)L + qb + m) * DH;
    Frag aq[4];
#pragma unroll
    for (int c = 0; c < 4; ++c) {
        aq[c].u[0] = ((const uint4*)(qg + 32 * c + 8 * hi))[0];
        aq[c].u[1] = ((const uint4*)(qg + 32 * c + 16 + 8 * hi))[0];
    }

    v8f ctx[8];
#pragma unroll
    for (int j = 0; j < 8; ++j)
#pragma unroll
        for (int r = 0; r < 8; ++r) ctx[j][r] = 0.f;
    float mrun[8], lrun[8];
#pragma unroll
    for (int r = 0; r < 8; ++r) { mrun[r] = -3.0e38f; lrun[r] = 0.f; }

    const __bf16* kg = Kr + (size_t)bh * L * DH;
    const __bf16* vg = Vt + (size_t)bh * DH * (size_t)L;
    __bf16* PsmW = Psm + wid * 16 * PSTR;

    const int ktEnd = qt * 64 + 64;
    for (int kt = 0; kt < ktEnd; kt += 32) {
        // ---- TDM-stage K tile (32 keys x 256B) and V tile (128 d x 64B) ----
        if (issuer) {
            tdm_load_2d(kg + (size_t)kt * DH, lds_addr_of(Ksm), 32, 32, 32, 5, 3);
            tdm_load_2d(vg + kt, lds_addr_of(Vsm), 8, 128, (unsigned)(L >> 2), 3, 3);
            __builtin_amdgcn_s_wait_tensorcnt(0);
        }
        __syncthreads();

        // ---- scores: S(16x32) = Q(16x128) @ K^T ----
        v8f S[2];
#pragma unroll
        for (int j2 = 0; j2 < 2; ++j2)
#pragma unroll
            for (int r = 0; r < 8; ++r) S[j2][r] = 0.f;
#pragma unroll
        for (int c = 0; c < 4; ++c) {
#pragma unroll
            for (int j2 = 0; j2 < 2; ++j2) {
                Frag kb;  // vec[e] = K[key=16*j2+m][32c + e + 8*hi] (e<8), [32c + e + 8 + 8*hi] (e>=8)
                const __bf16* kr = Ksm + (16 * j2 + m) * KSTR + 32 * c + 8 * hi;
                kb.u[0] = ((const uint4*)kr)[0];
                kb.u[1] = ((const uint4*)kr)[2];
                S[j2] = wmma_bf16(aq[c], kb, S[j2]);
            }
        }

        // ---- online softmax (rows live across the 16 lanes of a half-wave) ----
        float pv[2][8];
#pragma unroll
        for (int r = 0; r < 8; ++r) {
            int qrow = qb + r + 8 * hi;
            float s0 = S[0][r] * scale;
            float s1 = S[1][r] * scale;
            if (kt + m > qrow) s0 = -1.0e9f;
            if (kt + 16 + m > qrow) s1 = -1.0e9f;
            float mx = fmaxf(s0, s1);
#pragma unroll
            for (int off = 1; off < 16; off <<= 1)
                mx = fmaxf(mx, __shfl_xor(mx, off, 32));
            float mnew = fmaxf(mrun[r], mx);
            float sc = __expf(mrun[r] - mnew);
            float p0 = __expf(s0 - mnew);
            float p1 = __expf(s1 - mnew);
            float sum = p0 + p1;
#pragma unroll
            for (int off = 1; off < 16; off <<= 1)
                sum += __shfl_xor(sum, off, 32);
            lrun[r] = lrun[r] * sc + sum;
            mrun[r] = mnew;
#pragma unroll
            for (int j = 0; j < 8; ++j) ctx[j][r] *= sc;
            pv[0][r] = p0;
            pv[1][r] = p1;
        }

        // ---- re-layout P through per-wave LDS into A-fragment order ----
#pragma unroll
        for (int j2 = 0; j2 < 2; ++j2)
#pragma unroll
            for (int r = 0; r < 8; ++r)
                PsmW[(r + 8 * hi) * PSTR + 16 * j2 + m] = (__bf16)pv[j2][r];
        asm volatile("s_wait_dscnt 0" ::: "memory");

        Frag pf;
        const __bf16* pr = PsmW + m * PSTR;
        pf.u[0] = ((const uint4*)(pr + 8 * hi))[0];
        pf.u[1] = ((const uint4*)(pr + 16 + 8 * hi))[0];

        // ---- ctx(16x128) += P(16x32) @ V(32x128) ----
#pragma unroll
        for (int j = 0; j < 8; ++j) {
            Frag vb;  // vec[e] = V[key=e+8*hi (e<8), e+8+8*hi (e>=8)][d=16*j+m] = Vsm[d][key]
            const __bf16* vr = Vsm + (16 * j + m) * VSTR + 8 * hi;
            vb.u[0] = ((const uint4*)vr)[0];
            vb.u[1] = ((const uint4*)vr)[2];
            ctx[j] = wmma_bf16(pf, vb, ctx[j]);
        }
        __syncthreads();   // all reads done before next TDM issue / combine
    }

    // ---- differential combine: out = ctx0/l0 - lam * ctx1/l1 ----
    if (p == 1) {
        const float* lp = lam + (size_t)bh * L + qb;
#pragma unroll
        for (int j = 0; j < 8; ++j)
#pragma unroll
            for (int r = 0; r < 8; ++r) {
                int mm = r + 8 * hi;
                float lv = lp[mm];
                Csm[(wsub * 16 + mm) * 128 + 16 * j + m] = ctx[j][r] / lrun[r] * lv;
            }
    }
    __syncthreads();
    if (p == 0) {
#pragma unroll
        for (int j = 0; j < 8; ++j)
#pragma unroll
            for (int r = 0; r < 8; ++r) {
                int mm = r + 8 * hi;
                float o = ctx[j][r] / lrun[r] - Csm[(wsub * 16 + mm) * 128 + 16 * j + m];
                int l = qb + mm;
                Aout[((size_t)b * L + l) * 2048 + h * 128 + 16 * j + m] = (__bf16)o;
            }
    }
}

// ---------------------------------------------------------------------------
// Host launcher
// ---------------------------------------------------------------------------
extern "C" void kernel_launch(void* const* d_in, const int* in_sizes, int n_in,
                              void* d_out, int out_size, void* d_ws, size_t ws_size,
                              hipStream_t stream) {
    (void)in_sizes; (void)n_in; (void)out_size; (void)ws_size;
    const float* x  = (const float*)d_in[0];
    const float* Wq = (const float*)d_in[1];
    const float* Wk = (const float*)d_in[2];
    const float* Wv = (const float*)d_in[3];
    const float* Wl = (const float*)d_in[4];
    const float* bl = (const float*)d_in[5];
    const float* Wo = (const float*)d_in[6];
    float* out = (float*)d_out;

    constexpr int B = 2, L = 2048, D = 2048, NH = 16, DH = 128;
    const size_t ML = (size_t)B * L;  // 4096 rows

    char* ws = (char*)d_ws;
    size_t off = 0;
    auto alloc = [&](size_t bytes) -> void* {
        void* pp = ws + off;
        off += (bytes + 255) & ~(size_t)255;
        return pp;
    };
    __bf16* xb   = (__bf16*)alloc(ML * D * 2);
    __bf16* WqT  = (__bf16*)alloc((size_t)2 * D * D * 2);  // [2D][D]
    __bf16* WkT  = (__bf16*)alloc((size_t)D * D * 2);      // [D][D]
    __bf16* WvT  = (__bf16*)alloc((size_t)D * D * 2);
    __bf16* WoT  = (__bf16*)alloc((size_t)D * D * 2);
    float*  cosT = (float*)alloc((size_t)L * DH * 4);
    float*  sinT = (float*)alloc((size_t)L * DH * 4);
    __bf16* Qr   = (__bf16*)alloc(ML * 2 * D * 2);   // [b][2h+p][l][128]
    __bf16* Kr   = (__bf16*)alloc(ML * D * 2);       // [b][h][l][128]
    __bf16* Vtp  = (__bf16*)alloc(ML * D * 2);       // [b][h][d][l]
    float*  lamb = (float*)alloc(ML * NH * 4);       // [b][h][l]
    __bf16* Aout = (__bf16*)alloc(ML * D * 2);       // [b][l][D]

    cvt_bf16_kernel<<<(int)((ML * D + 255) / 256), 256, 0, stream>>>(x, xb, ML * D);
    // convert + transpose weights: W[K=D][N] -> Wt[N][K]
    cvtT_kernel<<<dim3(2 * D / 32, D / 32), 256, 0, stream>>>(Wq, WqT, D, 2 * D);
    cvtT_kernel<<<dim3(D / 32, D / 32), 256, 0, stream>>>(Wk, WkT, D, D);
    cvtT_kernel<<<dim3(D / 32, D / 32), 256, 0, stream>>>(Wv, WvT, D, D);
    cvtT_kernel<<<dim3(D / 32, D / 32), 256, 0, stream>>>(Wo, WoT, D, D);

    rope_table_kernel<<<(L * DH + 255) / 256, 256, 0, stream>>>(cosT, sinT, L, DH);

    // Q projection + RoPE:  (4096 x 2048) x Wt(4096 x 2048)
    gemm_bf16_wmma<EPI_ROPE><<<dim3(2 * D / 128, (int)(ML / 128)), 256, 0, stream>>>(
        xb, WqT, (int)ML, 2 * D, D, nullptr, Qr, cosT, sinT, L);
    // K projection + RoPE
    gemm_bf16_wmma<EPI_ROPE><<<dim3(D / 128, (int)(ML / 128)), 256, 0, stream>>>(
        xb, WkT, (int)ML, D, D, nullptr, Kr, cosT, sinT, L);
    // V projection, transposed store
    gemm_bf16_wmma<EPI_VT><<<dim3(D / 128, (int)(ML / 128)), 256, 0, stream>>>(
        xb, WvT, (int)ML, D, D, nullptr, Vtp, nullptr, nullptr, L);

    lam_kernel<<<(int)ML, 256, 0, stream>>>(x, Wl, bl, lamb, L, D, NH);

    attn_kernel<<<dim3(L / 64, B * NH), 256, 0, stream>>>(Qr, Kr, Vtp, lamb, Aout, L);

    // Output projection -> fp32 out
    gemm_bf16_wmma<EPI_PLAIN><<<dim3(D / 128, (int)(ML / 128)), 256, 0, stream>>>(
        Aout, WoT, (int)ML, D, D, out, nullptr, nullptr, nullptr, L);
}
